// NCMMLLoss_45981919871001
// MI455X (gfx1250) — compile-verified, matrix operand order checked by reading, not run on hardware
//
#include <hip/hip_runtime.h>
#include <hip/hip_bf16.h>

// Problem constants (from reference): N=16384, C=2048, D=128
#define NN 16384
#define CC 2048
#define DD 128

typedef float v8f  __attribute__((ext_vector_type(8)));
typedef __bf16 v8bf  __attribute__((ext_vector_type(8)));
typedef __bf16 v16bf __attribute__((ext_vector_type(16)));

union bfrag {
  v16bf v;
  v8bf  h[2];
};

// ---------------------------------------------------------------------------
// Workspace layout (bytes):
//   [0,       8K)   sq_m      : CC floats
//   [8K,     12K)   partial   : NN/16 floats
//   [16K,    80K)   sq_f      : NN floats
//   [80K, 80K+4M)   feat_hi   : NN*DD bf16
//   ... feat_lo (4M), mean_hi (512K), mean_lo (512K)
// ---------------------------------------------------------------------------
#define WS_SQM   0
#define WS_PART  (8u * 1024u)
#define WS_SQF   (16u * 1024u)
#define WS_FHI   (80u * 1024u)
#define WS_FLO   (WS_FHI + (size_t)NN * DD * 2)
#define WS_MHI   (WS_FLO + (size_t)NN * DD * 2)
#define WS_MLO   (WS_MHI + (size_t)CC * DD * 2)

// ---------------------------------------------------------------------------
// Pack kernels: split fp32 -> bf16 hi + bf16 lo (x ~= hi + lo), and row norms.
// ---------------------------------------------------------------------------
__global__ __launch_bounds__(256) void ncmm_pack_feat_kernel(
    const float* __restrict__ feat, __bf16* __restrict__ hi,
    __bf16* __restrict__ lo, float* __restrict__ sq_f) {
  int i = blockIdx.x * blockDim.x + threadIdx.x;
  if (i >= NN) return;
  const float* row = feat + (size_t)i * DD;
  __bf16* rh = hi + (size_t)i * DD;
  __bf16* rl = lo + (size_t)i * DD;
  float s = 0.f;
#pragma unroll 4
  for (int j = 0; j < DD; ++j) {
    float x = row[j];
    s += x * x;
    __bf16 h = (__bf16)x;            // RNE
    __bf16 l = (__bf16)(x - (float)h);
    rh[j] = h;
    rl[j] = l;
  }
  sq_f[i] = s;
}

__global__ __launch_bounds__(256) void ncmm_pack_mean_kernel(
    const float* __restrict__ means, __bf16* __restrict__ hi,
    __bf16* __restrict__ lo, float* __restrict__ sq_m) {
  int c = blockIdx.x * blockDim.x + threadIdx.x;
  if (c >= CC) return;
  const float* row = means + (size_t)c * DD;
  __bf16* rh = hi + (size_t)c * DD;
  __bf16* rl = lo + (size_t)c * DD;
  float s = 0.f;
#pragma unroll 4
  for (int j = 0; j < DD; ++j) {
    float x = row[j];
    s += x * x;
    __bf16 h = (__bf16)x;
    __bf16 l = (__bf16)(x - (float)h);
    rh[j] = h;
    rl[j] = l;
  }
  sq_m[c] = s;
}

// ---------------------------------------------------------------------------
// Main kernel: bf16x3-split distance GEMM (v_wmma_f32_16x16x32_bf16) fused
// with streaming log-sum-exp and target NLL. One wave32 per 16-row band.
// ---------------------------------------------------------------------------
__global__ __launch_bounds__(128) void ncmm_main_kernel(
    const float*  __restrict__ feat,      // [N, D] f32 (exact, for target dot)
    const float*  __restrict__ means,     // [C, D] f32 (exact, for target dot)
    const int*    __restrict__ targets,   // [N]
    const __bf16* __restrict__ feat_hi,   // [N, D]
    const __bf16* __restrict__ feat_lo,   // [N, D]
    const __bf16* __restrict__ mean_hi,   // [C, D]
    const __bf16* __restrict__ mean_lo,   // [C, D]
    const float*  __restrict__ sq_f,      // [N]
    const float*  __restrict__ sq_m,      // [C]
    float*        __restrict__ partial) { // [N/16]
  const int lane = threadIdx.x & 31;
  const int wave = threadIdx.x >> 5;
  const int band = blockIdx.x * 4 + wave;  // 256 blocks * 4 waves = 1024 bands
  const int m0   = band * 16;

  const int r  = lane & 15;  // A-row / B-col index for this lane
  const int hh = lane >> 4;  // lane half
  const int g0 = 8 * hh;     // K-group base within each 32-K chunk

  // ---- A panel (16x128) as bf16 hi/lo frags, resident in registers ----
  // ISA 16-bit A 16x32 layout: lane holds K = {g0..g0+7} U {g0+16..g0+23}
  bfrag a_hi[4], a_lo[4];
  {
    const __bf16* ah = feat_hi + (size_t)(m0 + r) * DD;
    const __bf16* al = feat_lo + (size_t)(m0 + r) * DD;
#pragma unroll
    for (int c = 0; c < 4; ++c) {
      a_hi[c].h[0] = *(const v8bf*)(ah + 32 * c + g0);
      a_hi[c].h[1] = *(const v8bf*)(ah + 32 * c + g0 + 16);
      a_lo[c].h[0] = *(const v8bf*)(al + 32 * c + g0);
      a_lo[c].h[1] = *(const v8bf*)(al + 32 * c + g0 + 16);
    }
  }

  // ---- exact ||f||^2 for own row, and for rows owned by each acc VGPR ----
  float sqf = sq_f[m0 + r];
  float sqf_row[8];
#pragma unroll
  for (int v = 0; v < 8; ++v)
    sqf_row[v] = __shfl(sqf, v + 8 * hh, 32);  // row m = v + 8*hh

  // ---- streaming log-sum-exp state per (lane, v) slot ----
  float run_max[8], run_sum[8];
#pragma unroll
  for (int v = 0; v < 8; ++v) { run_max[v] = -INFINITY; run_sum[v] = 0.f; }

  // ---- sweep all C columns in 16-wide tiles ----
  for (int n0 = 0; n0 < CC; n0 += 16) {
    const __bf16* bh = mean_hi + (size_t)(n0 + r) * DD;
    const __bf16* bl = mean_lo + (size_t)(n0 + r) * DD;
    bfrag b_hi[4], b_lo[4];
#pragma unroll
    for (int c = 0; c < 4; ++c) {
      b_hi[c].h[0] = *(const v8bf*)(bh + 32 * c + g0);
      b_hi[c].h[1] = *(const v8bf*)(bh + 32 * c + g0 + 16);
      b_lo[c].h[0] = *(const v8bf*)(bl + 32 * c + g0);
      b_lo[c].h[1] = *(const v8bf*)(bl + 32 * c + g0 + 16);
    }

    // bf16x3 split: cross ~= hi*hi + hi*lo + lo*hi, accumulated in f32
    v8f acc = {};
#pragma unroll
    for (int c = 0; c < 4; ++c) {
      acc = __builtin_amdgcn_wmma_f32_16x16x32_bf16(
          false, a_hi[c].v, false, b_hi[c].v, (short)0, acc, false, false);
      acc = __builtin_amdgcn_wmma_f32_16x16x32_bf16(
          false, a_hi[c].v, false, b_lo[c].v, (short)0, acc, false, false);
      acc = __builtin_amdgcn_wmma_f32_16x16x32_bf16(
          false, a_lo[c].v, false, b_hi[c].v, (short)0, acc, false, false);
    }

    float sm = sq_m[n0 + r];  // this lane's column norm
#pragma unroll
    for (int v = 0; v < 8; ++v) {
      float d2    = sqf_row[v] + sm - 2.f * acc[v];
      float logit = -__builtin_sqrtf(fmaxf(d2, 0.f));
      float nm    = fmaxf(run_max[v], logit);
      run_sum[v]  = run_sum[v] * __expf(run_max[v] - nm) + __expf(logit - nm);
      run_max[v]  = nm;
    }
  }

  // ---- merge the 16 column-subsets within each lane half (rows fixed) ----
  float logZ[8];
#pragma unroll
  for (int v = 0; v < 8; ++v) {
    float mx = run_max[v], sume = run_sum[v];
#pragma unroll
    for (int off = 1; off < 16; off <<= 1) {  // xor 1,2,4,8 stays in-half
      float om = __shfl_xor(mx, off, 32);
      float os = __shfl_xor(sume, off, 32);
      float nm = fmaxf(mx, om);
      sume = sume * __expf(mx - nm) + os * __expf(om - nm);
      mx = nm;
    }
    logZ[v] = mx + __logf(sume);
  }

  // ---- logZ for this lane's own row r = lane&15 ----
  float lzsel = logZ[0];
#pragma unroll
  for (int v = 1; v < 8; ++v)
    if ((lane & 7) == v) lzsel = logZ[v];
  int srcl   = (lane & 7) | (((lane >> 3) & 1) << 4);
  float lz_r = __shfl(lzsel, srcl, 32);

  // ---- exact target distance for row r (fp32 dot, half row per lane) ----
  int t = targets[m0 + r];
  const float4* fr = (const float4*)(feat + (size_t)(m0 + r) * DD) + 16 * hh;
  const float4* mr = (const float4*)(means + (size_t)t * DD) + 16 * hh;
  float dp = 0.f;
#pragma unroll
  for (int i = 0; i < 16; ++i) {
    float4 x = fr[i];
    float4 y = mr[i];
    dp += x.x * y.x + x.y * y.y + x.z * y.z + x.w * y.w;
  }
  dp += __shfl_xor(dp, 16, 32);

  float d2t    = sqf + sq_m[t] - 2.f * dp;
  float dist_t = __builtin_sqrtf(fmaxf(d2t, 0.f));
  float nll    = dist_t + lz_r;  // -(logit_t - logZ)

  // ---- sum the 16 rows (lanes 0..15 only; other half duplicates) ----
  float contrib = (lane < 16) ? nll : 0.f;
#pragma unroll
  for (int off = 1; off < 32; off <<= 1)
    contrib += __shfl_xor(contrib, off, 32);
  if (lane == 0) partial[band] = contrib;
}

// ---------------------------------------------------------------------------
// Deterministic reduction of 1024 band partials -> mean NLL
// ---------------------------------------------------------------------------
__global__ __launch_bounds__(256) void ncmm_reduce_kernel(
    const float* __restrict__ partial, float* __restrict__ out) {
  __shared__ float sh[256];
  int t = threadIdx.x;
  float v = partial[t] + partial[t + 256] + partial[t + 512] + partial[t + 768];
  sh[t] = v;
  __syncthreads();
  for (int s = 128; s > 0; s >>= 1) {
    if (t < s) sh[t] += sh[t + s];
    __syncthreads();
  }
  if (t == 0) out[0] = sh[0] * (1.f / (float)NN);
}

// ---------------------------------------------------------------------------
extern "C" void kernel_launch(void* const* d_in, const int* in_sizes, int n_in,
                              void* d_out, int out_size, void* d_ws, size_t ws_size,
                              hipStream_t stream) {
  const float* feat    = (const float*)d_in[0];  // [16384,128] f32
  const float* means   = (const float*)d_in[1];  // [2048,128]  f32
  const int*   targets = (const int*)d_in[2];    // [16384]     i32
  float* out = (float*)d_out;

  char* ws = (char*)d_ws;
  float*  sq_m    = (float*)(ws + WS_SQM);
  float*  partial = (float*)(ws + WS_PART);
  float*  sq_f    = (float*)(ws + WS_SQF);
  __bf16* feat_hi = (__bf16*)(ws + WS_FHI);
  __bf16* feat_lo = (__bf16*)(ws + WS_FLO);
  __bf16* mean_hi = (__bf16*)(ws + WS_MHI);
  __bf16* mean_lo = (__bf16*)(ws + WS_MLO);

  ncmm_pack_feat_kernel<<<NN / 256, 256, 0, stream>>>(feat, feat_hi, feat_lo,
                                                      sq_f);
  ncmm_pack_mean_kernel<<<CC / 256, 256, 0, stream>>>(means, mean_hi, mean_lo,
                                                      sq_m);
  ncmm_main_kernel<<<NN / 16 / 4, 128, 0, stream>>>(
      feat, means, targets, feat_hi, feat_lo, mean_hi, mean_lo, sq_f, sq_m,
      partial);
  ncmm_reduce_kernel<<<1, 256, 0, stream>>>(partial, out);
}
